// NuGraphCore_54829552501219
// MI455X (gfx1250) — matrix operand order repaired and checked by reading, not compiled, not run on hardware
//
#include <hip/hip_runtime.h>
#include <hip/hip_bf16.h>
#include <math.h>

// ---------------------------------------------------------------------------
// Problem constants (match reference setup_inputs)
// ---------------------------------------------------------------------------
#define NPL   50000
#define FPL   128
#define NSPN  30000
#define FNX   64
#define NEV   16
#define FIN   64

typedef __attribute__((ext_vector_type(16))) __bf16 bf16x16;
typedef __attribute__((ext_vector_type(8)))  __bf16 bf16x8;
typedef __attribute__((ext_vector_type(8)))  float  f32x8;

// ---------------------------------------------------------------------------
// Device helpers
// ---------------------------------------------------------------------------
__device__ __forceinline__ float mishf(float x) {
    // x * tanh(softplus(x)); large-x saturates correctly through inf
    return x * tanhf(log1pf(expf(x)));
}

__device__ __forceinline__ void atomicMaxF(float* addr, float v) {
    // float max via monotone int/uint punning (lowers to global_atomic_max/min)
    if (v >= 0.0f) atomicMax((int*)addr, __float_as_int(v));
    else           atomicMin((unsigned int*)addr, __float_as_uint(v));
}

// ---------------------------------------------------------------------------
// Utility kernels
// ---------------------------------------------------------------------------
__global__ void fill_u32_kernel(unsigned int* __restrict__ p, unsigned int v, int n) {
    int i = blockIdx.x * blockDim.x + threadIdx.x;
    if (i < n) p[i] = v;
}

// W [K,N] f32 row-major  ->  Wt [N,K] bf16 row-major (B^T for WMMA B-fragment)
__global__ void wconv_t_kernel(const float* __restrict__ W, __bf16* __restrict__ Wt,
                               int K, int N) {
    int i = blockIdx.x * blockDim.x + threadIdx.x;
    if (i >= K * N) return;
    int k = i / N, n = i % N;
    Wt[(long)n * K + k] = (__bf16)W[i];
}

// ---------------------------------------------------------------------------
// Edge stage 1: attention score + segment max.  One wave32 per edge.
// score = x_dst[d] . We[0:Fd] + x_src[s] . We[Fd:Fd+Fs] + be
// ---------------------------------------------------------------------------
__global__ void edge_score_max_kernel(const float* __restrict__ xsrc,
                                      const float* __restrict__ xdst,
                                      const int*   __restrict__ es,
                                      const int*   __restrict__ ed,
                                      const float* __restrict__ We,
                                      const float* __restrict__ be,
                                      float* __restrict__ abuf,
                                      float* __restrict__ maxb,
                                      int E, int Fs, int Fd) {
    int wid  = (blockIdx.x * blockDim.x + threadIdx.x) >> 5;
    int lane = threadIdx.x & 31;
    if (wid >= E) return;
    int s = es[wid], d = ed[wid];
    const float* xi = xdst + (long)d * Fd;
    const float* xj = xsrc + (long)s * Fs;
    float acc = 0.0f;
    for (int f = lane; f < Fd; f += 32) acc += xi[f] * We[f];
    for (int f = lane; f < Fs; f += 32) acc += xj[f] * We[Fd + f];
    #pragma unroll
    for (int off = 16; off; off >>= 1) acc += __shfl_xor(acc, off, 32);
    float a = 1.0f / (1.0f + expf(-(acc + be[0])));
    if (lane == 0) abuf[wid] = a;
    float* mrow = maxb + (long)d * Fs;
    for (int f = lane; f < Fs; f += 32) atomicMaxF(&mrow[f], a * xj[f]);
}

// ---------------------------------------------------------------------------
// Edge stage 2: exp-shifted numerator & denominator sums (native f32 atomics)
// ---------------------------------------------------------------------------
__global__ void edge_expsum_kernel(const float* __restrict__ xsrc,
                                   const int*   __restrict__ es,
                                   const int*   __restrict__ ed,
                                   const float* __restrict__ abuf,
                                   const float* __restrict__ maxb,
                                   float* __restrict__ numb,
                                   float* __restrict__ denb,
                                   int E, int Fs) {
    int wid  = (blockIdx.x * blockDim.x + threadIdx.x) >> 5;
    int lane = threadIdx.x & 31;
    if (wid >= E) return;
    int s = es[wid], d = ed[wid];
    float a = abuf[wid];
    const float* xj = xsrc + (long)s * Fs;
    long base = (long)d * Fs;
    for (int f = lane; f < Fs; f += 32) {
        float msg = a * xj[f];
        float ex  = expf(msg - maxb[base + f]);
        atomicAdd(&denb[base + f], ex);
        atomicAdd(&numb[base + f], ex * msg);
    }
}

// ---------------------------------------------------------------------------
// Build h = [aggr | x_dst] in bf16 (aggr = num/den, 0 for empty segments)
// ---------------------------------------------------------------------------
__global__ void pack_h_kernel(const float* __restrict__ numb,
                              const float* __restrict__ denb,
                              const float* __restrict__ xdst,
                              __bf16* __restrict__ h,
                              int Nd, int Fs, int Fd) {
    int K = Fs + Fd;
    int i = blockIdx.x * blockDim.x + threadIdx.x;
    if (i >= Nd * K) return;
    int dnode = i / K, c = i % K;
    float v;
    if (c < Fs) {
        float dd = denb[(long)dnode * Fs + c];
        v = (dd > 0.0f) ? numb[(long)dnode * Fs + c] / dd : 0.0f;
    } else {
        v = xdst[(long)dnode * Fd + (c - Fs)];
    }
    h[i] = (__bf16)v;
}

// ---------------------------------------------------------------------------
// WMMA GEMM + bias + mish.  One wave per 16x64 C slab: 4 accumulators give
// 4 independent wmma chains (hides XDL latency) and 4x A-fragment reuse.
//   A  : [M,K] bf16 row-major (activations)
//   Bt : [N,K] bf16 row-major (weights pre-transposed)
// A-fragment (16-bit A 16x32): m = lane%16; lanes<16 hold K kb..kb+15,
// lanes>=16 hold K kb+16..kb+31                -> two contiguous 16B loads.
// B-fragment (16-bit B 32x16): n = lane%16; lanes<16 hold K kb..kb+15,
// lanes>=16 hold K kb+16..kb+31                -> one contiguous 32B load.
// C/D f32 16x16: VGPR r -> row r + (lane<16?0:8), col lane%16.
// ---------------------------------------------------------------------------
__global__ void wmma_gemm_mish_kernel(const __bf16* __restrict__ A,
                                      const __bf16* __restrict__ Bt,
                                      const float*  __restrict__ bias,
                                      int M, int N, int K,
                                      float* __restrict__ outF,
                                      __bf16* __restrict__ outB,
                                      int accum) {
    int wid  = blockIdx.x * (blockDim.x >> 5) + (threadIdx.x >> 5);
    int lane = threadIdx.x & 31;
    int tilesN = N >> 6;                 // 64 columns per wave
    int tiles  = (M >> 4) * tilesN;
    if (wid >= tiles) return;            // whole wave retires: EXEC stays all-1s
    int mt = wid / tilesN, nt = wid % tilesN;
    int half = lane >> 4;                // 0 | 1
    int lm   = lane & 15;

    const __bf16* arow = A  + (long)(mt * 16 + lm) * K + half * 16;
    const __bf16* brow = Bt + (long)(nt * 64 + lm) * K + half * 16;
    const long bq = (long)16 * K;        // next 16-column group of Bt

    f32x8 c0 = {}, c1 = {}, c2 = {}, c3 = {};
    for (int kb = 0; kb < K; kb += 32) {
        bf16x8  a0 = *(const bf16x8*)(arow + kb);
        bf16x8  a1 = *(const bf16x8*)(arow + kb + 8);
        bf16x16 b0 = *(const bf16x16*)(brow + kb);
        bf16x16 b1 = *(const bf16x16*)(brow + bq + kb);
        bf16x16 b2 = *(const bf16x16*)(brow + 2 * bq + kb);
        bf16x16 b3 = *(const bf16x16*)(brow + 3 * bq + kb);
        if (kb + 32 < K)                 // WGP-scope prefetch of next A slab
            __builtin_prefetch(arow + kb + 32, 0, 3);
        bf16x16 afr;
        #pragma unroll
        for (int i = 0; i < 8; ++i) { afr[i] = a0[i]; afr[i + 8] = a1[i]; }
        c0 = __builtin_amdgcn_wmma_f32_16x16x32_bf16(false, afr, false, b0, (short)0, c0, false, false);
        c1 = __builtin_amdgcn_wmma_f32_16x16x32_bf16(false, afr, false, b1, (short)0, c1, false, false);
        c2 = __builtin_amdgcn_wmma_f32_16x16x32_bf16(false, afr, false, b2, (short)0, c2, false, false);
        c3 = __builtin_amdgcn_wmma_f32_16x16x32_bf16(false, afr, false, b3, (short)0, c3, false, false);
    }

    f32x8 cq[4] = {c0, c1, c2, c3};
    #pragma unroll
    for (int q = 0; q < 4; ++q) {
        int col = nt * 64 + q * 16 + lm;
        float bv = bias[col];
        #pragma unroll
        for (int r = 0; r < 8; ++r) {
            int row = mt * 16 + r + half * 8;
            float v = mishf(cq[q][r] + bv);
            long o = (long)row * N + col;
            if (outB) outB[o] = (__bf16)v;
            else      outF[o] = accum ? (outF[o] + v) : v;
        }
    }
}

// ---------------------------------------------------------------------------
// Host-side block orchestration
// ---------------------------------------------------------------------------
struct BlockPrm { const float *We, *be, *W1, *b1, *W2, *b2; };

static void run_block(hipStream_t st,
                      const float* xsrc, const float* xdst,
                      const int* e, int E,
                      int Fs, int Fd, int dout, int Nd,
                      const BlockPrm& p,
                      float* out, int accum,
                      float* abuf, float* maxb, float* numb, float* denb,
                      __bf16* h, __bf16* h2, __bf16* w1t, __bf16* w2t) {
    const int* es = e;
    const int* ed = e + E;
    int K1 = Fs + Fd;

    int nms = Nd * Fs;
    fill_u32_kernel<<<(nms + 255) / 256, 256, 0, st>>>((unsigned int*)maxb, 0xFF800000u, nms);
    fill_u32_kernel<<<(nms + 255) / 256, 256, 0, st>>>((unsigned int*)numb, 0u, nms);
    fill_u32_kernel<<<(nms + 255) / 256, 256, 0, st>>>((unsigned int*)denb, 0u, nms);

    wconv_t_kernel<<<(K1 * dout + 255) / 256, 256, 0, st>>>(p.W1, w1t, K1, dout);
    wconv_t_kernel<<<(dout * dout + 255) / 256, 256, 0, st>>>(p.W2, w2t, dout, dout);

    int eb = (E + 7) / 8;                 // 8 waves (edges) per 256-thread block
    edge_score_max_kernel<<<eb, 256, 0, st>>>(xsrc, xdst, es, ed, p.We, p.be,
                                              abuf, maxb, E, Fs, Fd);
    edge_expsum_kernel<<<eb, 256, 0, st>>>(xsrc, es, ed, abuf, maxb,
                                           numb, denb, E, Fs);

    int nh = Nd * K1;
    pack_h_kernel<<<(nh + 255) / 256, 256, 0, st>>>(numb, denb, xdst, h, Nd, Fs, Fd);

    int tiles = (Nd / 16) * (dout / 64);  // one wave per 16x64 slab
    wmma_gemm_mish_kernel<<<(tiles + 7) / 8, 256, 0, st>>>(h,  w1t, p.b1, Nd, dout, K1,
                                                           (float*)nullptr, h2, 0);
    wmma_gemm_mish_kernel<<<(tiles + 7) / 8, 256, 0, st>>>(h2, w2t, p.b2, Nd, dout, dout,
                                                           out, (__bf16*)nullptr, accum);
}

// ---------------------------------------------------------------------------
// Entry point
// ---------------------------------------------------------------------------
extern "C" void kernel_launch(void* const* d_in, const int* in_sizes, int n_in,
                              void* d_out, int out_size, void* d_ws, size_t ws_size,
                              hipStream_t stream) {
    (void)n_in; (void)out_size; (void)ws_size;
    // ---- inputs (setup_inputs dict order) ----
    const float* p_u   = (const float*)d_in[0];
    const float* p_v   = (const float*)d_in[1];
    const float* p_y   = (const float*)d_in[2];
    const float* n_sp  = (const float*)d_in[3];
    const float* i_evt = (const float*)d_in[4];
    const int* e_plane[3] = {(const int*)d_in[5], (const int*)d_in[6], (const int*)d_in[7]};
    const int* e_up[3]    = {(const int*)d_in[8], (const int*)d_in[9], (const int*)d_in[10]};
    const int* e_down[3]  = {(const int*)d_in[11], (const int*)d_in[12], (const int*)d_in[13]};
    const int* e_in   = (const int*)d_in[14];
    const int* e_owns = (const int*)d_in[15];
    int E_plane[3] = {in_sizes[5] / 2, in_sizes[6] / 2, in_sizes[7] / 2};
    int E_up[3]    = {in_sizes[8] / 2, in_sizes[9] / 2, in_sizes[10] / 2};
    int E_down[3]  = {in_sizes[11] / 2, in_sizes[12] / 2, in_sizes[13] / 2};
    int E_in   = in_sizes[14] / 2;
    int E_owns = in_sizes[15] / 2;

    auto prm = [&](int base) {
        BlockPrm b;
        b.We = (const float*)d_in[base + 0]; b.be = (const float*)d_in[base + 1];
        b.W1 = (const float*)d_in[base + 2]; b.b1 = (const float*)d_in[base + 3];
        b.W2 = (const float*)d_in[base + 4]; b.b2 = (const float*)d_in[base + 5];
        return b;
    };
    BlockPrm plane_p = prm(16), up_p = prm(22), n2i_p = prm(28),
             i2n_p = prm(34), down_p = prm(40);

    // ---- workspace carve (all 256B-aligned) ----
    char* wsb = (char*)d_ws; size_t off = 0;
    auto carve = [&](size_t bytes) -> void* {
        void* r = wsb + off; off += (bytes + 255) & ~(size_t)255; return r;
    };
    float* pu = (float*)carve((size_t)NPL * FPL * 4);   // stage-1 plane features
    float* pv = (float*)carve((size_t)NPL * FPL * 4);
    float* py = (float*)carve((size_t)NPL * FPL * 4);
    float* n1 = (float*)carve((size_t)NSPN * FNX * 4);  // nexus after up-sum
    float* abuf = (float*)carve((size_t)512000 * 4);    // per-edge attention
    float* maxb = (float*)carve((size_t)NPL * FPL * 4); // segment max/num/den
    float* numb = (float*)carve((size_t)NPL * FPL * 4);
    float* denb = (float*)carve((size_t)NPL * FPL * 4);
    __bf16* h   = (__bf16*)carve((size_t)NPL * 256 * 2); // [Nd, Fs+Fd] activations
    __bf16* h2  = (__bf16*)carve((size_t)NPL * FPL * 2); // [Nd, dout] hidden
    __bf16* w1t = (__bf16*)carve((size_t)256 * 128 * 2);
    __bf16* w2t = (__bf16*)carve((size_t)128 * 128 * 2);

    // ---- output regions (return order: pu2, pv2, py2, n2, i1) ----
    float* o = (float*)d_out;
    float* o_pl2[3] = {o, o + (size_t)NPL * FPL, o + 2 * (size_t)NPL * FPL};
    float* o_n2 = o + 3 * (size_t)NPL * FPL;
    float* o_i1 = o_n2 + (size_t)NSPN * FNX;

    const float* Pin[3]  = {p_u, p_v, p_y};
    float*       Pout[3] = {pu, pv, py};

    // 1) intra-plane message passing (shared weights)
    for (int pl = 0; pl < 3; ++pl)
        run_block(stream, Pin[pl], Pin[pl], e_plane[pl], E_plane[pl],
                  FPL, FPL, FPL, NPL, plane_p, Pout[pl], 0,
                  abuf, maxb, numb, denb, h, h2, w1t, w2t);

    // 2) plane -> nexus, summed across planes (HeteroConv aggr='sum')
    for (int pl = 0; pl < 3; ++pl)
        run_block(stream, Pout[pl], n_sp, e_up[pl], E_up[pl],
                  FPL, FNX, FNX, NSPN, up_p, n1, (pl > 0) ? 1 : 0,
                  abuf, maxb, numb, denb, h, h2, w1t, w2t);

    // 3) nexus -> interaction
    run_block(stream, n1, i_evt, e_in, E_in,
              FNX, FIN, FIN, NEV, n2i_p, o_i1, 0,
              abuf, maxb, numb, denb, h, h2, w1t, w2t);

    // 4) interaction -> nexus
    run_block(stream, o_i1, n1, e_owns, E_owns,
              FIN, FNX, FNX, NSPN, i2n_p, o_n2, 0,
              abuf, maxb, numb, denb, h, h2, w1t, w2t);

    // 5) nexus -> plane (shared weights)
    for (int pl = 0; pl < 3; ++pl)
        run_block(stream, o_n2, Pout[pl], e_down[pl], E_down[pl],
                  FNX, FPL, FPL, NPL, down_p, o_pl2[pl], 0,
                  abuf, maxb, numb, denb, h, h2, w1t, w2t);
}
